// Classifier_62843961475695
// MI455X (gfx1250) — matrix operand order repaired and checked by reading, not compiled, not run on hardware
//
#include <hip/hip_runtime.h>
#include <hip/hip_bf16.h>

// ---------------------------------------------------------------------------
// Types / helpers
// ---------------------------------------------------------------------------
typedef __bf16 bf16_t;
typedef __bf16 v16bf __attribute__((ext_vector_type(16)));
typedef __bf16 v8bf  __attribute__((ext_vector_type(8)));
typedef float  v8f   __attribute__((ext_vector_type(8)));

#define WMMA_BF16(a, b, c) \
  __builtin_amdgcn_wmma_f32_16x16x32_bf16(false, (a), false, (b), (short)0, (c), false, false)

static __device__ __forceinline__ float sigm(float x) { return 1.f / (1.f + expf(-x)); }

// A-fragment (16x32 bf16, MxK) from row-major LDS [row][k].
// lane l<16 holds row=l, K={0..7}u{16..23}; lane>=16 holds K={8..15}u{24..31}.
static __device__ __forceinline__ v16bf frag_a(const bf16_t* __restrict__ base, int ld,
                                               int row, int kk, int hl) {
  const bf16_t* p = base + row * ld + kk + hl * 8;
  v8bf lo = *(const v8bf*)(p);
  v8bf hi = *(const v8bf*)(p + 16);
  v16bf r;
#pragma unroll
  for (int i = 0; i < 8; ++i) { r[i] = lo[i]; r[i + 8] = hi[i]; }
  return r;
}

// B-fragment (32x16 bf16, KxN) from row-major LDS [n][k].
// lanes 0-15: N=lane, K={0..7}u{16..23}; lanes 16-31: N=lane-16, K={8..15}u{24..31}.
static __device__ __forceinline__ v16bf frag_b(const bf16_t* __restrict__ base, int ld,
                                               int col, int kk, int hl) {
  const bf16_t* p = base + col * ld + kk + hl * 8;
  v8bf lo = *(const v8bf*)(p);
  v8bf hi = *(const v8bf*)(p + 16);
  v16bf r;
#pragma unroll
  for (int i = 0; i < 8; ++i) { r[i] = lo[i]; r[i + 8] = hi[i]; }
  return r;
}

// ---------------------------------------------------------------------------
// Generic WMMA GEMM:  Y[m,n] = sum_k X[m,k]*W[n,k] (+bias[n]) (+resid[m,n])
// BM=BN=128, BK=64, 256 threads = 8 waves (4 x 2), wave tile = 32x64.
// Double-buffered LDS; next tile is loaded to registers before compute so
// global latency overlaps WMMA.  Full-tile fast path avoids exec-masked loads.
// Requires Kd % 64 == 0 (true for all call sites). M,N guarded on slow path.
// Shared: 2 buffers * 2 * 128*72 bf16 = 73728 bytes (dynamic).
// ---------------------------------------------------------------------------
__global__ __launch_bounds__(256) void k_gemm(
    const float* __restrict__ X, int ldx,
    const float* __restrict__ W, int ldw,
    const float* __restrict__ bias,
    const float* __restrict__ resid,
    float* __restrict__ Y, int ldy,
    int M, int N, int Kd) {
  extern __shared__ char smem[];
  const int LDA = 72;  // 64 + 8 pad (bf16 elems)
  const int BUFSZ = 2 * 128 * LDA;     // A + B per buffer (bf16 elems)

  const int tid = threadIdx.x;
  const int wave = tid >> 5, lane = tid & 31;
  const int l15 = lane & 15, hl = lane >> 4;
  const int wm = wave >> 1, wn = wave & 1;
  const int m0 = blockIdx.x * 128, n0 = blockIdx.y * 128;
  const bool full = (m0 + 128 <= M) && (n0 + 128 <= N);

  float4 ra[8], rb[8];

  auto load_regs = [&](int k0) {
    if (full) {
#pragma unroll
      for (int it = 0; it < 8; ++it) {
        int slot = tid + it * 256;
        int r = slot >> 4;            // 16 float4 per row
        int c = (slot & 15) << 2;
        ra[it] = *(const float4*)(X + (size_t)(m0 + r) * ldx + k0 + c);
        rb[it] = *(const float4*)(W + (size_t)(n0 + r) * ldw + k0 + c);
      }
    } else {
#pragma unroll
      for (int it = 0; it < 8; ++it) {
        int slot = tid + it * 256;
        int r = slot >> 4;
        int c = (slot & 15) << 2;
        ra[it] = make_float4(0.f, 0.f, 0.f, 0.f);
        rb[it] = make_float4(0.f, 0.f, 0.f, 0.f);
        if (m0 + r < M) ra[it] = *(const float4*)(X + (size_t)(m0 + r) * ldx + k0 + c);
        if (n0 + r < N) rb[it] = *(const float4*)(W + (size_t)(n0 + r) * ldw + k0 + c);
      }
    }
  };

  auto store_lds = [&](int buf) {
    bf16_t* A = (bf16_t*)smem + buf * BUFSZ;
    bf16_t* B = A + 128 * LDA;
#pragma unroll
    for (int it = 0; it < 8; ++it) {
      int slot = tid + it * 256;
      int r = slot >> 4;
      int c = (slot & 15) << 2;
      bf16_t* da = A + r * LDA + c;
      da[0] = (bf16_t)ra[it].x; da[1] = (bf16_t)ra[it].y;
      da[2] = (bf16_t)ra[it].z; da[3] = (bf16_t)ra[it].w;
      bf16_t* db = B + r * LDA + c;
      db[0] = (bf16_t)rb[it].x; db[1] = (bf16_t)rb[it].y;
      db[2] = (bf16_t)rb[it].z; db[3] = (bf16_t)rb[it].w;
    }
  };

  v8f acc[2][4] = {};

  const int nk = Kd >> 6;
  load_regs(0);
  store_lds(0);
  __syncthreads();

  for (int t = 0; t < nk; ++t) {
    const int cur = t & 1;
    if (t + 1 < nk) load_regs((t + 1) << 6);   // overlap with compute below

    const bf16_t* As = (const bf16_t*)smem + cur * BUFSZ;
    const bf16_t* Bs = As + 128 * LDA;
#pragma unroll
    for (int kk = 0; kk < 64; kk += 32) {
      v16bf af[2];
#pragma unroll
      for (int mi = 0; mi < 2; ++mi)
        af[mi] = frag_a(As, LDA, wm * 32 + mi * 16 + l15, kk, hl);
#pragma unroll
      for (int ni = 0; ni < 4; ++ni) {
        v16bf bfr = frag_b(Bs, LDA, wn * 64 + ni * 16 + l15, kk, hl);
#pragma unroll
        for (int mi = 0; mi < 2; ++mi)
          acc[mi][ni] = WMMA_BF16(af[mi], bfr, acc[mi][ni]);
      }
    }
    if (t + 1 < nk) {
      __syncthreads();          // everyone done reading buffer (t+1)&1 from iter t-1
      store_lds((t + 1) & 1);
      __syncthreads();          // new tile visible before next iteration
    }
  }

  // store
#pragma unroll
  for (int mi = 0; mi < 2; ++mi) {
#pragma unroll
    for (int ni = 0; ni < 4; ++ni) {
      int col = n0 + wn * 64 + ni * 16 + l15;
      float bv = (bias && col < N) ? bias[col] : 0.f;
#pragma unroll
      for (int r = 0; r < 8; ++r) {
        int row = m0 + wm * 32 + mi * 16 + hl * 8 + r;
        if (row < M && col < N) {
          size_t off = (size_t)row * ldy + col;
          float v = acc[mi][ni][r] + bv;
          if (resid) v += resid[off];
          Y[off] = v;
        }
      }
    }
  }
}

// ---------------------------------------------------------------------------
// Embedding gather: x[row, :] = emb[tok[row], :]   (rows = B*S, D=512)
// ---------------------------------------------------------------------------
__global__ void k_embed(const int* __restrict__ tok, const float* __restrict__ emb,
                        float* __restrict__ x) {
  int i = blockIdx.x * 256 + threadIdx.x;     // over ROWS*128 float4
  int row = i >> 7;
  int c = (i & 127) << 2;
  float4 v = *(const float4*)(emb + (size_t)tok[row] * 512 + c);
  *(float4*)(x + (size_t)row * 512 + c) = v;
}

// ---------------------------------------------------------------------------
// LayerNorm (no bias): y = (x-mu)*rsqrt(var+eps)*w ; one block per row.
// ---------------------------------------------------------------------------
__global__ __launch_bounds__(256) void k_ln(const float* __restrict__ x, size_t xstride,
                                            const float* __restrict__ w,
                                            float* __restrict__ y, size_t ystride, int dim) {
  __shared__ float r1[256], r2[256];
  int row = blockIdx.x, tid = threadIdx.x;
  const float* xr = x + (size_t)row * xstride;
  float s = 0.f, s2 = 0.f;
  for (int i = tid; i < dim; i += 256) { float v = xr[i]; s += v; s2 += v * v; }
  r1[tid] = s; r2[tid] = s2;
  __syncthreads();
  for (int st = 128; st > 0; st >>= 1) {
    if (tid < st) { r1[tid] += r1[tid + st]; r2[tid] += r2[tid + st]; }
    __syncthreads();
  }
  float mu = r1[0] / dim;
  float var = r2[0] / dim - mu * mu;
  float rs = rsqrtf(var + 1e-5f);
  float* yr = y + (size_t)row * ystride;
  for (int i = tid; i < dim; i += 256) yr[i] = (xr[i] - mu) * rs * w[i];
}

// ---------------------------------------------------------------------------
// Causal depthwise conv (K=4) + optional SiLU.  x strided by ldin per row.
// ---------------------------------------------------------------------------
__global__ void k_conv(const float* __restrict__ x, int ldin,
                       const float* __restrict__ w, const float* __restrict__ bias,
                       float* __restrict__ y, int C, int dosilu) {
  int idx = blockIdx.x * 256 + threadIdx.x;
  int c = idx % C;
  int row = idx / C;
  int b = row >> 9, s = row & 511;
  float acc = bias[c];
#pragma unroll
  for (int j = 0; j < 4; ++j) {
    int sp = s - 3 + j;
    if (sp >= 0) acc += x[(size_t)(b * 512 + sp) * ldin + c] * w[c * 4 + j];
  }
  if (dosilu) acc = acc * sigm(acc);
  y[(size_t)row * C + c] = acc;
}

// ---------------------------------------------------------------------------
// Block-diag 4x4 projection: y[row, n*4+o] = sum_i x[row, n*4+i] * w[n,o,i]
// ---------------------------------------------------------------------------
__global__ void k_bproj(const float* __restrict__ x, int ldin,
                        const float* __restrict__ w, float* __restrict__ y) {
  int idx = blockIdx.x * 256 + threadIdx.x;   // over ROWS*1024
  int row = idx >> 10;
  int c = idx & 1023;
  int nb = c >> 2, o = c & 3;
  const float* xr = x + (size_t)row * ldin + (nb << 2);
  const float* wr = w + nb * 16 + o * 4;
  y[idx] = xr[0] * wr[0] + xr[1] * wr[1] + xr[2] * wr[2] + xr[3] * wr[3];
}

// ---------------------------------------------------------------------------
// mLSTM gate vectors: ig/fg[b,h,s] = qkv_row . w[h] + b[h]   (3072-dot, 4 heads)
// ---------------------------------------------------------------------------
__global__ __launch_bounds__(256) void k_mgates(
    const float* __restrict__ q, const float* __restrict__ k, const float* __restrict__ v,
    const float* __restrict__ igw, const float* __restrict__ igb,
    const float* __restrict__ fgw, const float* __restrict__ fgb,
    float* __restrict__ ig, float* __restrict__ fg) {
  __shared__ float red[256 * 8];
  int row = blockIdx.x, tid = threadIdx.x;
  int b = row >> 9, s = row & 511;
  float p[8] = {0.f, 0.f, 0.f, 0.f, 0.f, 0.f, 0.f, 0.f};
  for (int i = tid; i < 1024; i += 256) {
    float qv = q[(size_t)row * 1024 + i];
    float kv = k[(size_t)row * 1024 + i];
    float vv = v[(size_t)row * 1024 + i];
#pragma unroll
    for (int h = 0; h < 4; ++h) {
      p[h]     += qv * igw[h * 3072 + i] + kv * igw[h * 3072 + 1024 + i] + vv * igw[h * 3072 + 2048 + i];
      p[4 + h] += qv * fgw[h * 3072 + i] + kv * fgw[h * 3072 + 1024 + i] + vv * fgw[h * 3072 + 2048 + i];
    }
  }
#pragma unroll
  for (int h = 0; h < 8; ++h) red[tid * 8 + h] = p[h];
  __syncthreads();
  for (int st = 128; st > 0; st >>= 1) {
    if (tid < st) {
#pragma unroll
      for (int h = 0; h < 8; ++h) red[tid * 8 + h] += red[(tid + st) * 8 + h];
    }
    __syncthreads();
  }
  if (tid == 0) {
#pragma unroll
    for (int h = 0; h < 4; ++h) {
      ig[((size_t)b * 4 + h) * 512 + s] = red[h] + igb[h];
      fg[((size_t)b * 4 + h) * 512 + s] = red[4 + h] + fgb[h];
    }
  }
}

// ---------------------------------------------------------------------------
// mLSTM gate scans: cf = cumsum(logsigmoid(fg)); e = ig - cf;
// maxD = cf + runmax(ig - cf).   One block per (b,head); serial (512 steps).
// ---------------------------------------------------------------------------
__global__ void k_mprep(const float* __restrict__ fg, const float* __restrict__ ig,
                        float* __restrict__ cf, float* __restrict__ e,
                        float* __restrict__ mx) {
  if (threadIdx.x != 0) return;
  int bn = blockIdx.x;
  float csum = 0.f, rmax = -3.4e38f;
  for (int s = 0; s < 512; ++s) {
    float f = fg[(size_t)bn * 512 + s];
    float lf = (f >= 0.f) ? -log1pf(expf(-f)) : (f - log1pf(expf(f)));
    csum += lf;
    float a = ig[(size_t)bn * 512 + s] - csum;
    rmax = fmaxf(rmax, a);
    cf[(size_t)bn * 512 + s] = csum;
    e[(size_t)bn * 512 + s] = a;
    mx[(size_t)bn * 512 + s] = csum + rmax;
  }
}

// ---------------------------------------------------------------------------
// mLSTM cell (flash-attention style, single pass):
//   C[s,t] = (q.k)*dh^-.5 * exp(cf[s]-cf[t]+ig[t]-maxD[s]) for t<=s
//   out[s] = (sum_t C[s,t] v[t]) / (max(|rowsum|, exp(-maxD[s])) + 1e-6)
// grid (S/64, B*NH), 256 threads. q/k/v are (B*S,1024) with head col offset.
// hc out layout (B,NH,S,256).
// ---------------------------------------------------------------------------
__global__ __launch_bounds__(256) void k_mcell(
    const float* __restrict__ qg, const float* __restrict__ kg, const float* __restrict__ vg,
    const float* __restrict__ cf, const float* __restrict__ ev, const float* __restrict__ mx,
    float* __restrict__ hc) {
  extern __shared__ char smem[];
  bf16_t* Qs = (bf16_t*)smem;          // 64 x 264
  bf16_t* Ks = Qs + 64 * 264;          // 64 x 264
  bf16_t* Vt = Ks + 64 * 264;          // 256 x 72 (transposed V: [d][t])
  bf16_t* Sc = Vt + 256 * 72;          // 64 x 72  (bf16 scores)
  float* rowsum = (float*)(Sc + 64 * 72);
  float* basev = rowsum + 64;
  float* et = basev + 64;

  const int tid = threadIdx.x;
  const int wave = tid >> 5, lane = tid & 31;
  const int l15 = lane & 15, hl = lane >> 4;
  const int wm = wave >> 1, wn = wave & 1;
  const int bh = blockIdx.y;
  const int b = bh >> 2, n = bh & 3;
  const int s0 = blockIdx.x * 64;
  const size_t rowbase = (size_t)(b * 512) * 1024 + n * 256;
  const int bnS = bh * 512;

  // load Q tile 64x256
#pragma unroll
  for (int it = 0; it < 16; ++it) {
    int slot = tid + it * 256;
    int r = slot >> 6;
    int c = (slot & 63) << 2;
    float4 v = *(const float4*)(qg + rowbase + (size_t)(s0 + r) * 1024 + c);
    bf16_t* d = Qs + r * 264 + c;
    d[0] = (bf16_t)v.x; d[1] = (bf16_t)v.y; d[2] = (bf16_t)v.z; d[3] = (bf16_t)v.w;
  }
  if (tid < 64) {
    basev[tid] = cf[bnS + s0 + tid] - mx[bnS + s0 + tid];
    rowsum[tid] = 0.f;
  }

  v8f o[8] = {};
  const int ntt = (s0 >> 6) + 1;
  for (int tt = 0; tt < ntt; ++tt) {
    const int t0 = tt * 64;
    __syncthreads();
    // K tile 64x256
#pragma unroll
    for (int it = 0; it < 16; ++it) {
      int slot = tid + it * 256;
      int r = slot >> 6;
      int c = (slot & 63) << 2;
      float4 v = *(const float4*)(kg + rowbase + (size_t)(t0 + r) * 1024 + c);
      bf16_t* d = Ks + r * 264 + c;
      d[0] = (bf16_t)v.x; d[1] = (bf16_t)v.y; d[2] = (bf16_t)v.z; d[3] = (bf16_t)v.w;
    }
    // V tile transposed -> Vt[d][t]
#pragma unroll
    for (int it = 0; it < 16; ++it) {
      int slot = tid + it * 256;
      int r = slot >> 6;
      int c = (slot & 63) << 2;
      float4 v = *(const float4*)(vg + rowbase + (size_t)(t0 + r) * 1024 + c);
      Vt[(c + 0) * 72 + r] = (bf16_t)v.x;
      Vt[(c + 1) * 72 + r] = (bf16_t)v.y;
      Vt[(c + 2) * 72 + r] = (bf16_t)v.z;
      Vt[(c + 3) * 72 + r] = (bf16_t)v.w;
    }
    if (tid < 64) et[tid] = ev[bnS + t0 + tid];
    __syncthreads();

    // scores: wave computes 16 (rows, wm) x 32 (cols, wn) of the 64x64 tile
    v8f sc[2] = {};
#pragma unroll
    for (int kd = 0; kd < 256; kd += 32) {
      v16bf aq = frag_a(Qs, 264, wm * 16 + l15, kd, hl);
#pragma unroll
      for (int ni = 0; ni < 2; ++ni) {
        v16bf bk = frag_b(Ks, 264, wn * 32 + ni * 16 + l15, kd, hl);
        sc[ni] = WMMA_BF16(aq, bk, sc[ni]);
      }
    }
    // decay factor + causal mask + rowsum; re-stage as bf16 for scores @ V
#pragma unroll
    for (int ni = 0; ni < 2; ++ni) {
#pragma unroll
      for (int r = 0; r < 8; ++r) {
        int sr = wm * 16 + hl * 8 + r;
        int tc = wn * 32 + ni * 16 + l15;
        float fac = ((t0 + tc) <= (s0 + sr)) ? expf(basev[sr] + et[tc]) * 0.0625f : 0.f;
        float val = sc[ni][r] * fac;
        atomicAdd(&rowsum[sr], val);
        Sc[sr * 72 + tc] = (bf16_t)val;
      }
    }
    __syncthreads();

    // out += Sc(64x64) @ V(64x256); wave: 16 rows (wm) x 128 cols (wn)
#pragma unroll
    for (int kk = 0; kk < 64; kk += 32) {
      v16bf ac = frag_a(Sc, 72, wm * 16 + l15, kk, hl);
#pragma unroll
      for (int ni = 0; ni < 8; ++ni) {
        v16bf bv = frag_b(Vt, 72, wn * 128 + ni * 16 + l15, kk, hl);
        o[ni] = WMMA_BF16(ac, bv, o[ni]);
      }
    }
  }
  __syncthreads();
  // normalize and store
#pragma unroll
  for (int ni = 0; ni < 8; ++ni) {
    int dc = wn * 128 + ni * 16 + l15;
#pragma unroll
    for (int r = 0; r < 8; ++r) {
      int sr = wm * 16 + hl * 8 + r;
      float nrm = fmaxf(fabsf(rowsum[sr]), expf(-mx[bnS + s0 + sr]));
      hc[((size_t)bnS + s0 + sr) * 256 + dc] = o[ni][r] / (nrm + 1e-6f);
    }
  }
}

// ---------------------------------------------------------------------------
// Per-head group norm (LayerNorm per head) + optional residual add.
// in element (b,s,n,d) at b*sb + s*ss + n*sn + d. out (B*S, od).
// ---------------------------------------------------------------------------
__global__ __launch_bounds__(128) void k_mhnorm(
    const float* __restrict__ in, const float* __restrict__ w,
    const float* __restrict__ resid, float* __restrict__ out,
    int dh, size_t sb, size_t ss, size_t sn, int od) {
  __shared__ float r1[128], r2[128];
  int row = blockIdx.x, n = blockIdx.y, tid = threadIdx.x;
  int b = row >> 9, s = row & 511;
  const float* ir = in + (size_t)b * sb + (size_t)s * ss + (size_t)n * sn;
  float sm = 0.f, s2 = 0.f;
  for (int i = tid; i < dh; i += 128) { float v = ir[i]; sm += v; s2 += v * v; }
  r1[tid] = sm; r2[tid] = s2;
  __syncthreads();
  for (int st = 64; st > 0; st >>= 1) {
    if (tid < st) { r1[tid] += r1[tid + st]; r2[tid] += r2[tid + st]; }
    __syncthreads();
  }
  float mu = r1[0] / dh;
  float var = r2[0] / dh - mu * mu;
  float rs = rsqrtf(var + 1e-5f);
  for (int i = tid; i < dh; i += 128) {
    size_t off = (size_t)row * od + n * dh + i;
    float v = (ir[i] - mu) * rs * w[n * dh + i];
    if (resid) v += resid[off];
    out[off] = v;
  }
}

// ---------------------------------------------------------------------------
// mLSTM output gate: o = (hn + skip*xc) * silu(z)  (z = up[:,1024:])
// ---------------------------------------------------------------------------
__global__ void k_gateout(const float* __restrict__ hn, const float* __restrict__ xc,
                          const float* __restrict__ up, const float* __restrict__ skip,
                          float* __restrict__ o) {
  int idx = blockIdx.x * 256 + threadIdx.x;  // over ROWS*1024
  int c = idx & 1023;
  int row = idx >> 10;
  float z = up[(size_t)row * 2048 + 1024 + c];
  o[idx] = (hn[idx] + skip[c] * xc[idx]) * (z * sigm(z));
}

// ---------------------------------------------------------------------------
// sLSTM recurrent scan. One block (256 thr) per (b,head). R (4x128x128) in LDS.
// gx layout (B,S,4,NH,128); hs layout (B,S,NH,128).
// ---------------------------------------------------------------------------
__global__ __launch_bounds__(256) void k_sscan(
    const float* __restrict__ gx, const float* __restrict__ R,
    const float* __restrict__ bias, float* __restrict__ hs) {
  extern __shared__ char smem[];
  float* ldR = (float*)smem;             // [512][132]
  float* gl = ldR + 512 * 132;           // 512
  float* hT = gl + 512;                  // 128 x4 states
  float* cT = hT + 128;
  float* nT = cT + 128;
  float* mT = nT + 128;
  float* bb = mT + 128;                  // 512
  int tid = threadIdx.x;
  int bh = blockIdx.x;
  int b = bh >> 2, n = bh & 3;

#pragma unroll 4
  for (int it = 0; it < 256; ++it) {
    int idx = tid + it * 256;            // 0..65535
    int g = idx >> 14;
    int rem = idx & 16383;
    int k2 = rem >> 7, d = rem & 127;
    ldR[(g * 128 + k2) * 132 + d] = R[(((size_t)g * 4 + n) * 128 + k2) * 128 + d];
  }
  if (tid < 128) { hT[tid] = 0.f; cT[tid] = 0.f; nT[tid] = 0.f; mT[tid] = 0.f; }
  {
    int o1 = tid, o2 = tid + 256;
    bb[o1] = bias[((o1 >> 7) * 4 + n) * 128 + (o1 & 127)];
    bb[o2] = bias[((o2 >> 7) * 4 + n) * 128 + (o2 & 127)];
  }
  __syncthreads();

  for (int s = 0; s < 512; ++s) {
#pragma unroll
    for (int pass = 0; pass < 2; ++pass) {
      int o = tid + pass * 256;          // o = g*128 + k
      int g = o >> 7, k2 = o & 127;
      float acc = gx[((size_t)(b * 512 + s) * 4 + g) * 512 + n * 128 + k2] + bb[o];
      const float* rr = ldR + (size_t)o * 132;
#pragma unroll 4
      for (int d = 0; d < 128; ++d) acc += hT[d] * rr[d];
      gl[o] = acc;
    }
    __syncthreads();
    if (tid < 128) {
      int k2 = tid;
      float it_ = gl[k2], ft = gl[128 + k2], zt = gl[256 + k2], ot = gl[384 + k2];
      float mn = fmaxf(ft + mT[k2], it_);
      float ia = expf(it_ - mn);
      float fa = expf(ft + mT[k2] - mn);
      float cn = fa * cT[k2] + ia * tanhf(zt);
      float nn = fa * nT[k2] + ia;
      float hv = sigm(ot) * cn / nn;
      cT[k2] = cn; nT[k2] = nn; mT[k2] = mn; hT[k2] = hv;
      hs[((size_t)(b * 512 + s) * 4 + n) * 128 + k2] = hv;
    }
    __syncthreads();
  }
}

// ---------------------------------------------------------------------------
// FF: f = gelu(gate) * up  (ffo rows of 1408 = [gate 704 | up 704])
// ---------------------------------------------------------------------------
__global__ void k_gelu_mul(const float* __restrict__ ffo, float* __restrict__ out) {
  int idx = blockIdx.x * 256 + threadIdx.x;  // over ROWS*704
  int row = idx / 704;
  int j = idx % 704;
  float g = ffo[(size_t)row * 1408 + j];
  float u = ffo[(size_t)row * 1408 + 704 + j];
  out[idx] = 0.5f * g * (1.f + erff(g * 0.70710678f)) * u;
}

// ---------------------------------------------------------------------------
// Host orchestration
// ---------------------------------------------------------------------------
extern "C" void kernel_launch(void* const* d_in, const int* in_sizes, int n_in,
                              void* d_out, int out_size, void* d_ws, size_t ws_size,
                              hipStream_t stream) {
  (void)in_sizes; (void)n_in; (void)out_size; (void)ws_size;
  const int* tok = (const int*)d_in[0];
  const float* emb = (const float*)d_in[1];
  const float* m_ln_w = (const float*)d_in[2];
  const float* m_up_w = (const float*)d_in[3];
  const float* m_conv_w = (const float*)d_in[4];
  const float* m_conv_b = (const float*)d_in[5];
  const float* m_q_w = (const float*)d_in[6];
  const float* m_k_w = (const float*)d_in[7];
  const float* m_v_w = (const float*)d_in[8];
  const float* m_ig_w = (const float*)d_in[9];
  const float* m_ig_b = (const float*)d_in[10];
  const float* m_fg_w = (const float*)d_in[11];
  const float* m_fg_b = (const float*)d_in[12];
  const float* m_gn_w = (const float*)d_in[13];
  const float* m_skip = (const float*)d_in[14];
  const float* m_down_w = (const float*)d_in[15];
  const float* s_ln_w = (const float*)d_in[16];
  const float* s_conv_w = (const float*)d_in[17];
  const float* s_conv_b = (const float*)d_in[18];
  const float* s_wx = (const float*)d_in[19];
  const float* s_r = (const float*)d_in[20];
  const float* s_b = (const float*)d_in[21];
  const float* s_gn_w = (const float*)d_in[22];
  const float* s_ln2_w = (const float*)d_in[23];
  const float* s_ff_up_w = (const float*)d_in[24];
  const float* s_ff_down_w = (const float*)d_in[25];
  const float* pb_ln_w = (const float*)d_in[26];
  const float* fc_w = (const float*)d_in[27];
  const float* fc_b = (const float*)d_in[28];
  float* out = (float*)d_out;

  const int ROWS = 8 * 512;  // 4096
  float* x = (float*)d_ws;
  float* hln = x + (size_t)ROWS * 512;
  float* up = hln + (size_t)ROWS * 512;     // 2048 wide (also gx / ffup out)
  float* xc = up + (size_t)ROWS * 2048;     // 1024 wide (also yc / gelu buf)
  float* qb = xc + (size_t)ROWS * 1024;     // 1024 wide (also hs)
  float* kb = qb + (size_t)ROWS * 1024;
  float* vb = kb + (size_t)ROWS * 1024;
  float* hc = vb + (size_t)ROWS * 1024;
  float* hn = hc + (size_t)ROWS * 1024;
  float* ob = hn + (size_t)ROWS * 1024;
  float* ig = ob + (size_t)ROWS * 1024;
  float* fg = ig + 16384;
  float* cfb = fg + 16384;
  float* eb = cfb + 16384;
  float* mxb = eb + 16384;
  float* xlast = mxb + 16384;

  const size_t GSM = 2 * 2 * 128 * 72 * sizeof(bf16_t);                      // 73728
  const size_t CSM = (2 * 64 * 264 + 256 * 72 + 64 * 72) * sizeof(bf16_t)
                     + (64 * 3) * sizeof(float);                             // 114432
  const size_t SSM = (512 * 132 + 512 + 4 * 128 + 512) * sizeof(float);      // 276480

  k_embed<<<ROWS * 128 / 256, 256, 0, stream>>>(tok, emb, x);

  int im = 0, isl = 0;
  for (int blk = 0; blk < 11; ++blk) {
    bool isS = (blk == 1 || blk == 4 || blk == 7 || blk == 10);
    if (!isS) {
      const float* lnw = m_ln_w + (size_t)im * 512;
      const float* upw = m_up_w + (size_t)im * 2048 * 512;
      const float* cw = m_conv_w + (size_t)im * 1024 * 4;
      const float* cb = m_conv_b + (size_t)im * 1024;
      const float* qw = m_q_w + (size_t)im * 256 * 16;
      const float* kw = m_k_w + (size_t)im * 256 * 16;
      const float* vw = m_v_w + (size_t)im * 256 * 16;
      const float* igw = m_ig_w + (size_t)im * 4 * 3072;
      const float* igb = m_ig_b + (size_t)im * 4;
      const float* fgw = m_fg_w + (size_t)im * 4 * 3072;
      const float* fgb = m_fg_b + (size_t)im * 4;
      const float* gnw = m_gn_w + (size_t)im * 1024;
      const float* skp = m_skip + (size_t)im * 1024;
      const float* dww = m_down_w + (size_t)im * 512 * 1024;

      k_ln<<<ROWS, 256, 0, stream>>>(x, 512, lnw, hln, 512, 512);
      { dim3 g(32, 16);
        k_gemm<<<g, 256, GSM, stream>>>(hln, 512, upw, 512, nullptr, nullptr,
                                        up, 2048, ROWS, 2048, 512); }
      k_conv<<<ROWS * 1024 / 256, 256, 0, stream>>>(up, 2048, cw, cb, xc, 1024, 1);
      k_bproj<<<ROWS * 1024 / 256, 256, 0, stream>>>(xc, 1024, qw, qb);
      k_bproj<<<ROWS * 1024 / 256, 256, 0, stream>>>(xc, 1024, kw, kb);
      k_bproj<<<ROWS * 1024 / 256, 256, 0, stream>>>(up, 2048, vw, vb);
      k_mgates<<<ROWS, 256, 0, stream>>>(qb, kb, vb, igw, igb, fgw, fgb, ig, fg);
      k_mprep<<<32, 32, 0, stream>>>(fg, ig, cfb, eb, mxb);
      { dim3 g(8, 32);
        k_mcell<<<g, 256, CSM, stream>>>(qb, kb, vb, cfb, eb, mxb, hc); }
      { dim3 g(ROWS, 4);
        k_mhnorm<<<g, 128, 0, stream>>>(hc, gnw, nullptr, hn,
                                        256, (size_t)4 * 512 * 256, 256,
                                        (size_t)512 * 256, 1024); }
      k_gateout<<<ROWS * 1024 / 256, 256, 0, stream>>>(hn, xc, up, skp, ob);
      { dim3 g(32, 4);
        k_gemm<<<g, 256, GSM, stream>>>(ob, 1024, dww, 1024, nullptr, x,
                                        x, 512, ROWS, 512, 1024); }
      ++im;
    } else {
      const float* lnw = s_ln_w + (size_t)isl * 512;
      const float* cw = s_conv_w + (size_t)isl * 512 * 4;
      const float* cb = s_conv_b + (size_t)isl * 512;
      const float* wx = s_wx + (size_t)isl * 4 * 4 * 128 * 128;
      const float* Rp = s_r + (size_t)isl * 4 * 4 * 128 * 128;
      const float* bp = s_b + (size_t)isl * 4 * 4 * 128;
      const float* gnw = s_gn_w + (size_t)isl * 512;
      const float* ln2w = s_ln2_w + (size_t)isl * 512;
      const float* ffu = s_ff_up_w + (size_t)isl * 1408 * 512;
      const float* ffd = s_ff_down_w + (size_t)isl * 512 * 704;

      k_ln<<<ROWS, 256, 0, stream>>>(x, 512, lnw, hln, 512, 512);
      k_conv<<<ROWS * 512 / 256, 256, 0, stream>>>(hln, 512, cw, cb, xc, 512, 1);
      // gate projections: gi,gf from yc(xc); gz,go from y(hln)
      for (int g4 = 0; g4 < 4; ++g4) {
        for (int n4 = 0; n4 < 4; ++n4) {
          const float* Xp = (g4 < 2 ? xc : hln) + n4 * 128;
          const float* Wp = wx + ((size_t)g4 * 4 + n4) * 128 * 128;
          float* Yp = up + (g4 * 4 + n4) * 128;
          dim3 g(32, 1);
          k_gemm<<<g, 256, GSM, stream>>>(Xp, 512, Wp, 128, nullptr, nullptr,
                                          Yp, 2048, ROWS, 128, 128);
        }
      }
      k_sscan<<<32, 256, SSM, stream>>>(up, Rp, bp, qb);
      { dim3 g(ROWS, 4);
        k_mhnorm<<<g, 128, 0, stream>>>(qb, gnw, x, x,
                                        128, (size_t)512 * 512, 512, 128, 512); }
      k_ln<<<ROWS, 256, 0, stream>>>(x, 512, ln2w, hln, 512, 512);
      { dim3 g(32, 11);
        k_gemm<<<g, 256, GSM, stream>>>(hln, 512, ffu, 512, nullptr, nullptr,
                                        up, 1408, ROWS, 1408, 512); }
      k_gelu_mul<<<ROWS * 704 / 256, 256, 0, stream>>>(up, xc);
      { dim3 g(32, 4);
        k_gemm<<<g, 256, GSM, stream>>>(xc, 704, ffd, 704, nullptr, x,
                                        x, 512, ROWS, 512, 704); }
      ++isl;
    }
  }

  // final LN on last token of each batch row, then classifier GEMM
  k_ln<<<8, 256, 0, stream>>>(x + (size_t)511 * 512, (size_t)512 * 512,
                              pb_ln_w, xlast, 512, 512);
  { dim3 g(1, 4);
    k_gemm<<<g, 256, GSM, stream>>>(xlast, 512, fc_w, 512, fc_b, nullptr,
                                    out, 512, 8, 512, 512); }
}
